// NonLocalBlock_21766894256860
// MI455X (gfx1250) — matrix operand order repaired and checked by reading, not compiled, not run on hardware
//
#include <hip/hip_runtime.h>

typedef __attribute__((ext_vector_type(16))) _Float16 v16h;
typedef __attribute__((ext_vector_type(8)))  _Float16 v8h;
typedef __attribute__((ext_vector_type(8)))  float    v8f;
typedef _Float16 half_t;

#define BATCH 8
#define CHN   256
#define IC    128
#define NPOS  4096

static __device__ __forceinline__ v8f wmma_f16(v16h a, v16h b, v8f c) {
  // (neg_a, A, neg_b, B, c_mod, C, reuse_a, reuse_b)
  return __builtin_amdgcn_wmma_f32_16x16x32_f16(false, a, false, b, (short)0, c,
                                                false, false);
}

static __device__ __forceinline__ v16h cat8(v8h lo, v8h hi) {
  return __builtin_shufflevector(lo, hi, 0, 1, 2, 3, 4, 5, 6, 7,
                                 8, 9, 10, 11, 12, 13, 14, 15);
}

// ---------------------------------------------------------------------------
// Kernel 1: fused QKV projection.
// [384x256] x [256x4096] per batch with f16 WMMA (K=32 chunks).
// Rows 0..127 = theta -> qT[n][d], 128..255 = phi -> kT[n][d],
// 256..383 = g -> v[d][n].
// grid = (NPOS/16, 384/16, BATCH), block = 32 (one wave).
// ---------------------------------------------------------------------------
__global__ __launch_bounds__(32)
void nlb_proj_kernel(const float* __restrict__ x,
                     const float* __restrict__ g_w,  const float* __restrict__ g_b,
                     const float* __restrict__ th_w, const float* __restrict__ th_b,
                     const float* __restrict__ ph_w, const float* __restrict__ ph_b,
                     half_t* __restrict__ qT, half_t* __restrict__ kT,
                     half_t* __restrict__ v)
{
  const int lane = threadIdx.x;
  const int hl   = lane >> 4;   // half of wave (k-range selector)
  const int lr   = lane & 15;   // row/col within tile
  const int n0   = blockIdx.x * 16;
  const int row0 = blockIdx.y * 16;
  const int b    = blockIdx.z;

  const float* xb = x + (size_t)b * CHN * NPOS;

  // A-fragment row for this lane (tiles never straddle the 128-row boundaries)
  const int grow = row0 + lr;
  const float* wrow;
  if (grow < IC)            wrow = th_w + (size_t)grow * CHN;
  else if (grow < 2 * IC)   wrow = ph_w + (size_t)(grow - IC) * CHN;
  else                      wrow = g_w  + (size_t)(grow - 2 * IC) * CHN;

  v8f acc = {};
  for (int kk = 0; kk < CHN; kk += 32) {
    v16h a, bf;
    const int ka0 = kk + hl * 16;       // A: e<8  -> k = hl*16+e
    const int ka1 = kk + hl * 16 + 8;   // A: e>=8 -> k = hl*16+e
#pragma unroll
    for (int e = 0; e < 8; ++e) a[e]     = (half_t)wrow[ka0 + e];
#pragma unroll
    for (int e = 0; e < 8; ++e) a[8 + e] = (half_t)wrow[ka1 + e];
    const int kb0 = kk + hl * 16;       // B: e -> k = hl*16+e, col = lr
#pragma unroll
    for (int e = 0; e < 16; ++e)
      bf[e] = (half_t)xb[(size_t)(kb0 + e) * NPOS + n0 + lr];
    acc = wmma_f16(a, bf, acc);
  }

  // D element j -> row M = row0 + j + 8*hl, col n = n0 + lr
#pragma unroll
  for (int j = 0; j < 8; ++j) {
    const int r = row0 + j + 8 * hl;
    const int n = n0 + lr;
    float bias; half_t* dst;
    if (r < IC) {
      bias = th_b[r];
      dst  = qT + ((size_t)b * NPOS + n) * IC + r;
    } else if (r < 2 * IC) {
      bias = ph_b[r - IC];
      dst  = kT + ((size_t)b * NPOS + n) * IC + (r - IC);
    } else {
      bias = g_b[r - 2 * IC];
      dst  = v + ((size_t)b * IC + (r - 2 * IC)) * NPOS + n;
    }
    *dst = (half_t)(acc[j] + bias);
  }
}

// ---------------------------------------------------------------------------
// Kernel 2: fused attention (scores -> softmax -> PV); S never hits memory.
// One wave per 16-query tile. Two passes over all 4096 keys:
//   pass 1: row max of S;  pass 2: P=exp(S-max), O += P*V^T, then O /= sum.
// All K/Q/V fragment loads are aligned 16B/32B vector loads.
// grid = (NPOS/16, BATCH), block = 32.
// ---------------------------------------------------------------------------
__global__ __launch_bounds__(32)
void nlb_attn_kernel(const half_t* __restrict__ qT, const half_t* __restrict__ kT,
                     const half_t* __restrict__ vv, half_t* __restrict__ y)
{
  __shared__ half_t pbuf[16][40];  // P staging; 80B rows keep 16B alignment

  const int lane  = threadIdx.x;
  const int hl    = lane >> 4;
  const int lr    = lane & 15;
  const int qtile = blockIdx.x;
  const int b     = blockIdx.y;

  const half_t* qb = qT + (size_t)b * NPOS * IC;
  const half_t* kb = kT + (size_t)b * NPOS * IC;
  const half_t* vb = vv + (size_t)b * IC * NPOS;

  // Q fragments (4 d-chunks of 32) stay in registers for both passes.
  v16h aq[4];
  {
    const half_t* qr = qb + (size_t)(qtile * 16 + lr) * IC;
#pragma unroll
    for (int c4 = 0; c4 < 4; ++c4) {
      const v8h lo = *(const v8h*)(qr + c4 * 32 + hl * 16);
      const v8h hi = *(const v8h*)(qr + c4 * 32 + hl * 16 + 8);
      aq[c4] = cat8(lo, hi);
    }
  }

  // ---- pass 1: running row max ----
  float m[8];
#pragma unroll
  for (int j = 0; j < 8; ++j) m[j] = -1e30f;

  for (int kt = 0; kt < NPOS / 16; ++kt) {
    const half_t* kr = kb + (size_t)(kt * 16 + lr) * IC;
    if (kt + 1 < NPOS / 16)
      __builtin_prefetch(kb + (size_t)((kt + 1) * 16 + lr) * IC + hl * 64, 0, 1);
    v8f s = {};
#pragma unroll
    for (int c4 = 0; c4 < 4; ++c4) {
      const v16h bk = *(const v16h*)(kr + c4 * 32 + hl * 16);
      s = wmma_f16(aq[c4], bk, s);
    }
#pragma unroll
    for (int j = 0; j < 8; ++j) m[j] = fmaxf(m[j], s[j]);
  }
  // reduce max within each 16-lane half (row M=j+8*hl lives in one half)
#pragma unroll
  for (int off = 1; off < 16; off <<= 1) {
#pragma unroll
    for (int j = 0; j < 8; ++j) m[j] = fmaxf(m[j], __shfl_xor(m[j], off, 16));
  }

  // ---- pass 2: exp + PV accumulation ----
  v8f o[8];
#pragma unroll
  for (int t = 0; t < 8; ++t) o[t] = (v8f){};
  float ssum[8];
#pragma unroll
  for (int j = 0; j < 8; ++j) ssum[j] = 0.f;

  for (int kc = 0; kc < NPOS / 32; ++kc) {
    const half_t* kr0 = kb + (size_t)(kc * 32 + lr) * IC;
    const half_t* kr1 = kb + (size_t)(kc * 32 + 16 + lr) * IC;
    if (kc + 1 < NPOS / 32)
      __builtin_prefetch(kb + (size_t)((kc + 1) * 32 + lane) * IC, 0, 1);
    v8f s0 = {}, s1 = {};
#pragma unroll
    for (int c4 = 0; c4 < 4; ++c4) {
      const v16h bk0 = *(const v16h*)(kr0 + c4 * 32 + hl * 16);
      const v16h bk1 = *(const v16h*)(kr1 + c4 * 32 + hl * 16);
      s0 = wmma_f16(aq[c4], bk0, s0);
      s1 = wmma_f16(aq[c4], bk1, s1);
    }
    // P = exp(S - max); stash into LDS in (q, key) order for A-frag reload
#pragma unroll
    for (int j = 0; j < 8; ++j) {
      const float p0 = __expf(s0[j] - m[j]);
      const float p1 = __expf(s1[j] - m[j]);
      ssum[j] += p0 + p1;
      pbuf[j + 8 * hl][lr]      = (half_t)p0;
      pbuf[j + 8 * hl][16 + lr] = (half_t)p1;
    }
    __syncthreads();
    const v8h plo = *(const v8h*)&pbuf[lr][hl * 16];
    const v8h phi = *(const v8h*)&pbuf[lr][hl * 16 + 8];
    const v16h ap = cat8(plo, phi);
    // O[q][i] += sum_k P[q][k] * V[k][i], V[k][i] = g_x[i][k]
#pragma unroll
    for (int t = 0; t < 8; ++t) {
      const v16h bvf =
          *(const v16h*)(vb + (size_t)(t * 16 + lr) * NPOS + kc * 32 + hl * 16);
      o[t] = wmma_f16(ap, bvf, o[t]);
    }
    __syncthreads();
  }

  // reduce exp-sums within half, normalize, store y as [n][i] f16
#pragma unroll
  for (int off = 1; off < 16; off <<= 1) {
#pragma unroll
    for (int j = 0; j < 8; ++j) ssum[j] += __shfl_xor(ssum[j], off, 16);
  }
#pragma unroll
  for (int t = 0; t < 8; ++t) {
#pragma unroll
    for (int j = 0; j < 8; ++j) {
      const int qn = qtile * 16 + j + 8 * hl;
      y[((size_t)b * NPOS + qn) * IC + t * 16 + lr] = (half_t)(o[t][j] / ssum[j]);
    }
  }
}

// ---------------------------------------------------------------------------
// Kernel 3: out = W_w @ y + W_b + x  (residual), fp32 store.
// grid = (NPOS/16, CHN/16, BATCH), block = 32.
// ---------------------------------------------------------------------------
__global__ __launch_bounds__(32)
void nlb_out_kernel(const float* __restrict__ x, const float* __restrict__ Ww,
                    const float* __restrict__ Wb, const half_t* __restrict__ y,
                    float* __restrict__ out)
{
  const int lane = threadIdx.x;
  const int hl   = lane >> 4;
  const int lr   = lane & 15;
  const int n0   = blockIdx.x * 16;
  const int c0   = blockIdx.y * 16;
  const int b    = blockIdx.z;

  const float*  wrow = Ww + (size_t)(c0 + lr) * IC;
  const half_t* yr   = y + ((size_t)b * NPOS + n0 + lr) * IC;

  v8f acc = {};
#pragma unroll
  for (int kk = 0; kk < IC; kk += 32) {
    v16h a;
#pragma unroll
    for (int e = 0; e < 8; ++e) a[e]     = (half_t)wrow[kk + hl * 16 + e];
#pragma unroll
    for (int e = 0; e < 8; ++e) a[8 + e] = (half_t)wrow[kk + hl * 16 + 8 + e];
    const v16h bf = *(const v16h*)(yr + kk + hl * 16);
    acc = wmma_f16(a, bf, acc);
  }
#pragma unroll
  for (int j = 0; j < 8; ++j) {
    const int c = c0 + j + 8 * hl;
    const int n = n0 + lr;
    const size_t idx = ((size_t)b * CHN + c) * NPOS + n;
    out[idx] = acc[j] + Wb[c] + x[idx];
  }
}

// ---------------------------------------------------------------------------
extern "C" void kernel_launch(void* const* d_in, const int* in_sizes, int n_in,
                              void* d_out, int out_size, void* d_ws, size_t ws_size,
                              hipStream_t stream) {
  const float* x    = (const float*)d_in[0];
  const float* g_w  = (const float*)d_in[1];
  const float* g_b  = (const float*)d_in[2];
  const float* th_w = (const float*)d_in[3];
  const float* th_b = (const float*)d_in[4];
  const float* ph_w = (const float*)d_in[5];
  const float* ph_b = (const float*)d_in[6];
  const float* W_w  = (const float*)d_in[7];
  const float* W_b  = (const float*)d_in[8];
  float* out = (float*)d_out;

  // f16 workspace: qT, kT (as [b][n][d]), v (as [b][d][n]), y (as [b][n][d])
  half_t* qT = (half_t*)d_ws;
  half_t* kT = qT + (size_t)BATCH * NPOS * IC;
  half_t* v  = kT + (size_t)BATCH * NPOS * IC;
  half_t* y  = v  + (size_t)BATCH * IC * NPOS;   // 32 MB total

  dim3 blk(32, 1, 1);
  nlb_proj_kernel<<<dim3(NPOS / 16, (3 * IC) / 16, BATCH), blk, 0, stream>>>(
      x, g_w, g_b, th_w, th_b, ph_w, ph_b, qT, kT, v);
  nlb_attn_kernel<<<dim3(NPOS / 16, BATCH), blk, 0, stream>>>(qT, kT, v, y);
  nlb_out_kernel<<<dim3(NPOS / 16, CHN / 16, BATCH), blk, 0, stream>>>(
      x, W_w, W_b, y, out);
}
